// MultiHeadAttention2d_68032281968841
// MI455X (gfx1250) — matrix operand order repaired and checked by reading, not compiled, not run on hardware
//
#include <hip/hip_runtime.h>
#include <hip/hip_bf16.h>

// MultiHeadAttention2d on gfx1250 (MI455X), wave32 + v_wmma_f32_16x16x32_bf16.
// Kernel 1: q/k/v/yp projections via WMMA, staging x/y tiles in LDS (bf16).
// Kernel 2: fused flash-style attention; K/V tiles double-buffered in LDS via
//           global_load_async_to_lds_b128 (ASYNCcnt) and shared by all 4 waves.

typedef __attribute__((ext_vector_type(16))) __bf16 v16bf;
typedef __attribute__((ext_vector_type(8)))  __bf16 v8bf;
typedef __attribute__((ext_vector_type(8)))  float  v8f;
typedef __attribute__((ext_vector_type(4)))  float  v4f;

__device__ __forceinline__ v8bf load8bf(const __bf16* p) { return *(const v8bf*)p; }

__device__ __forceinline__ v16bf join16(v8bf lo, v8bf hi) {
  v16bf r;
#pragma unroll
  for (int i = 0; i < 8; ++i) { r[i] = lo[i]; r[8 + i] = hi[i]; }
  return r;
}

// 8 contiguous f32 -> 8 bf16
__device__ __forceinline__ v8bf cvt8(const float* p) {
  v4f a = *(const v4f*)p;
  v4f b = *(const v4f*)(p + 4);
  v8bf r;
#pragma unroll
  for (int i = 0; i < 4; ++i) { r[i] = (__bf16)a[i]; r[4 + i] = (__bf16)b[i]; }
  return r;
}

__device__ __forceinline__ v8f wmma_bf16(v16bf a, v16bf b, v8f c) {
  return __builtin_amdgcn_wmma_f32_16x16x32_bf16(false, a, false, b, (short)0, c,
                                                 false, false);
}

// Async global->LDS DMA (16B). lds_off = LDS byte offset (low 32b of generic ptr).
__device__ __forceinline__ void async16(unsigned lds_off, unsigned long long gaddr) {
  asm volatile("global_load_async_to_lds_b128 %0, %1, off"
               :: "v"(lds_off), "v"(gaddr) : "memory");
}

__device__ __forceinline__ void wait_async0() {
#if __has_builtin(__builtin_amdgcn_s_wait_asynccnt)
  __builtin_amdgcn_s_wait_asynccnt(0);
#else
  asm volatile("s_wait_asynccnt 0x0" ::: "memory");
#endif
}

#define NTOK 4096
#define XS_STRIDE 88   // 64 + pad, 176 B rows (16B-aligned) for ds_load_b128
#define PST 48         // P tile stride (96 B rows, 16B-aligned)
#define KST 72         // K tile stride in elems (144 B rows, 16B-aligned)
#define VST 40         // V tile stride in elems (80 B rows, 16B-aligned)

// ---------------------------------------------------------------------------
// Kernel 1: projections. grid = B * (N/64) = 128 blocks of 128 threads
// (4 waves, each owns one 16-wide n sub-tile).
// Outputs: qT,kT : bf16 (b,h,n,64) ; vws : bf16 (b,h,64,n) ; ypws : f32 (b,64,n)
// ---------------------------------------------------------------------------
__global__ __launch_bounds__(128)
void mha_proj_kernel(const float* __restrict__ x,  const float* __restrict__ y,
                     const float* __restrict__ Wq, const float* __restrict__ bq,
                     const float* __restrict__ Wk, const float* __restrict__ bk,
                     const float* __restrict__ Wv, const float* __restrict__ bv,
                     const float* __restrict__ Wp,
                     __bf16* __restrict__ qT, __bf16* __restrict__ kT,
                     __bf16* __restrict__ vws, float* __restrict__ ypws) {
  __shared__ __bf16 xs[64 * XS_STRIDE];   // (n_local, c), c contiguous
  __shared__ __bf16 ys[64 * XS_STRIDE];

  const int t  = threadIdx.x;
  const int b  = blockIdx.x >> 6;
  const int n0 = (blockIdx.x & 63) << 6;  // 64 tokens per block

  // Transpose-load x,y: global (c, n) -> LDS (n, c), f32 -> bf16.
  for (int idx = t; idx < 64 * 64; idx += 128) {
    const int c = idx >> 6, j = idx & 63;
    xs[j * XS_STRIDE + c] = (__bf16)x[(b * 64 + c) * NTOK + n0 + j];
    ys[j * XS_STRIDE + c] = (__bf16)y[(b * 64 + c) * NTOK + n0 + j];
  }
  __syncthreads();

  const int w    = t >> 5;      // wave id -> n sub-tile
  const int lane = t & 31;
  const int li   = lane & 15;
  const int hi   = lane >> 4;   // half-wave select
  const int nl   = w * 16;
  const int nrow = nl + li;     // local token index for this lane
  const int n_g  = n0 + nrow;   // global token index

  // B fragments of X (K=c, N=n) for q,k : 16 contiguous c per lane.
  v16bf bx0, bx1;
  {
    const __bf16* r = &xs[nrow * XS_STRIDE + hi * 16];
    bx0 = join16(load8bf(r), load8bf(r + 8));
    bx1 = join16(load8bf(r + 32), load8bf(r + 40));
  }
  // A fragments of Y^T (M=n, K=c) for v,yp.
  v16bf ay0, ay1;
  {
    const __bf16* r = &ys[nrow * XS_STRIDE];
    ay0 = join16(load8bf(r + hi * 16),      load8bf(r + hi * 16 + 8));
    ay1 = join16(load8bf(r + 32 + hi * 16), load8bf(r + 40 + hi * 16));
  }

  // ---- q, k : D[M=o, N=n] = W(16o x 32c) * X(32c x 16n), store (n,d) bf16 ----
#pragma unroll 1
  for (int pass = 0; pass < 2; ++pass) {
    const float* W    = pass ? Wk : Wq;
    const float* bias = pass ? bk : bq;
    __bf16*      dst  = pass ? kT : qT;
#pragma unroll 1
    for (int o0 = 0; o0 < 256; o0 += 16) {
      const int orow = o0 + li;
      const float* wr = W + orow * 64;
      v16bf aw0 = join16(cvt8(wr + hi * 16),      cvt8(wr + hi * 16 + 8));
      v16bf aw1 = join16(cvt8(wr + 32 + hi * 16), cvt8(wr + 40 + hi * 16));
      v8f acc = {};
      acc = wmma_bf16(aw0, bx0, acc);
      acc = wmma_bf16(aw1, bx1, acc);
      const float* bp = bias + o0 + hi * 8;   // rows o = o0 + hi*8 + r
      v8bf outp;
#pragma unroll
      for (int r = 0; r < 8; ++r) outp[r] = (__bf16)(acc[r] + bp[r]);
      const int h  = o0 >> 6;
      const int d0 = (o0 & 63) + hi * 8;
      *(v8bf*)&dst[((size_t)(b * 4 + h) * NTOK + n_g) * 64 + d0] = outp;
    }
  }

  // ---- v : D[M=n, N=o] = Y^T(16n x 32c) * Wv^T(32c x 16o), store (d,n) bf16 ----
#pragma unroll 1
  for (int o0 = 0; o0 < 256; o0 += 16) {
    const int ocol = o0 + li;
    const float* wr = Wv + ocol * 64 + hi * 16;
    v16bf bw0 = join16(cvt8(wr),      cvt8(wr + 8));
    v16bf bw1 = join16(cvt8(wr + 32), cvt8(wr + 40));
    v8f acc = {};
    acc = wmma_bf16(ay0, bw0, acc);
    acc = wmma_bf16(ay1, bw1, acc);
    const float bvv = bv[ocol];
    v8bf outp;
#pragma unroll
    for (int r = 0; r < 8; ++r) outp[r] = (__bf16)(acc[r] + bvv);
    const int h = ocol >> 6, d = ocol & 63;
    *(v8bf*)&vws[((size_t)(b * 4 + h) * 64 + d) * NTOK + n0 + nl + hi * 8] = outp;
  }

  // ---- yp : like v, Wp (64x64), no bias, fp32 output (d,n) ----
#pragma unroll 1
  for (int o0 = 0; o0 < 64; o0 += 16) {
    const int ocol = o0 + li;
    const float* wr = Wp + ocol * 64 + hi * 16;
    v16bf bw0 = join16(cvt8(wr),      cvt8(wr + 8));
    v16bf bw1 = join16(cvt8(wr + 32), cvt8(wr + 40));
    v8f acc = {};
    acc = wmma_bf16(ay0, bw0, acc);
    acc = wmma_bf16(ay1, bw1, acc);
    float* op = &ypws[((size_t)(b * 64) + ocol) * NTOK + n0 + nl + hi * 8];
    v4f lo = {acc[0], acc[1], acc[2], acc[3]};
    v4f hi4 = {acc[4], acc[5], acc[6], acc[7]};
    *(v4f*)op = lo;
    *(v4f*)(op + 4) = hi4;
  }
}

// ---------------------------------------------------------------------------
// Kernel 2: fused attention. grid = B*nh*(N/64) = 512 blocks of 128 threads;
// each wave owns 16 query rows; the block streams all 4096 keys in 32-wide
// tiles, staged once per block into LDS with async DMA and double buffering.
// ---------------------------------------------------------------------------
__global__ __launch_bounds__(128)
void mha_attn_kernel(const __bf16* __restrict__ qT, const __bf16* __restrict__ kT,
                     const __bf16* __restrict__ vws, const float* __restrict__ ypws,
                     const float* __restrict__ gamma, float* __restrict__ out) {
  __shared__ __bf16 Kt[2][32 * KST];     // staged K tile (m, d), padded rows
  __shared__ __bf16 Vt[2][64 * VST];     // staged V tile (d, m), padded rows
  __shared__ __bf16 Ptile[4][16 * PST];  // per-wave exp(S) tile (n, m)
  __shared__ float  ls[4][16];           // per-wave row sums

  const int t = threadIdx.x, w = t >> 5, lane = t & 31;
  const int li = lane & 15, hi = lane >> 4;
  const int blk = blockIdx.x;
  const int b = blk >> 8;
  const int h = (blk >> 6) & 3;
  const int nt = blk & 63;
  const int bh = b * 4 + h;
  const int nbase = nt * 64 + w * 16;

  const __bf16* qrow  = qT + ((size_t)bh * NTOK + nbase) * 64;
  const __bf16* kbase = kT + (size_t)bh * NTOK * 64;
  const __bf16* vbase = vws + (size_t)bh * 64 * NTOK;

  // Q A-fragments (M=n, K=d), d = 0..31 and 32..63.
  const __bf16* qr = qrow + li * 64;
  const v16bf qa0 = join16(load8bf(qr + hi * 16),      load8bf(qr + hi * 16 + 8));
  const v16bf qa1 = join16(load8bf(qr + 32 + hi * 16), load8bf(qr + 40 + hi * 16));

  // Per-thread DMA slices: K = 32 rows x 128 B, V = 64 rows x 64 B.
  const int krow = t >> 2, kq = t & 3;   // K: 4 x 32 B chunks per row
  const int vrow = t >> 1, vh = t & 1;   // V: 2 x 32 B chunks per row
  const unsigned long long kg0 =
      (unsigned long long)(size_t)kbase + (size_t)krow * 128 + (size_t)kq * 32;
  const unsigned long long vg0 =
      (unsigned long long)(size_t)vbase + (size_t)vrow * (NTOK * 2) + (size_t)vh * 32;
  unsigned kl_off[2], vl_off[2];
#pragma unroll
  for (int bf = 0; bf < 2; ++bf) {
    kl_off[bf] = (unsigned)(size_t)&Kt[bf][krow * KST] + kq * 32;
    vl_off[bf] = (unsigned)(size_t)&Vt[bf][vrow * VST] + vh * 32;
  }

  auto issue_tile = [&](int bf, int m0) {
    const unsigned long long kg = kg0 + (size_t)m0 * 128;  // 128 B per key row
    const unsigned long long vg = vg0 + (size_t)m0 * 2;    // bf16 m offset
    async16(kl_off[bf],      kg);
    async16(kl_off[bf] + 16, kg + 16);
    async16(vl_off[bf],      vg);
    async16(vl_off[bf] + 16, vg + 16);
  };

  v8f accd[4];
#pragma unroll
  for (int i = 0; i < 4; ++i) { v8f z = {}; accd[i] = z; }
  v8f lpart = {};
  __bf16* Pw = Ptile[w];

  issue_tile(0, 0);
  wait_async0();
  __syncthreads();

#pragma unroll 1
  for (int m0 = 0; m0 < NTOK; m0 += 32) {
    const int cur = (m0 >> 5) & 1;
    if (m0 + 32 < NTOK) {
      issue_tile(cur ^ 1, m0 + 32);
      // distance-2 warmup of L2 from HBM on the first pass
      __builtin_prefetch(kbase + (size_t)(m0 + 64 + lane) * 64, 0, 1);
    }

    // K B-fragments (K=d, N=m) from LDS, two 16-wide key sub-tiles.
    const __bf16* kl0 = &Kt[cur][li * KST + hi * 16];
    const __bf16* kl1 = &Kt[cur][(16 + li) * KST + hi * 16];
    v16bf kb0 = join16(load8bf(kl0),      load8bf(kl0 + 8));
    v16bf kb1 = join16(load8bf(kl0 + 32), load8bf(kl0 + 40));
    v16bf kb2 = join16(load8bf(kl1),      load8bf(kl1 + 8));
    v16bf kb3 = join16(load8bf(kl1 + 32), load8bf(kl1 + 40));

    v8f s0 = {}, s1 = {};
    s0 = wmma_bf16(qa0, kb0, s0);
    s0 = wmma_bf16(qa1, kb1, s0);
    s1 = wmma_bf16(qa0, kb2, s1);
    s1 = wmma_bf16(qa1, kb3, s1);

    // exp + row-sum partials + stage P tile in LDS ((n, m) layout).
#pragma unroll
    for (int r = 0; r < 8; ++r) {
      const float e0 = __expf(s0[r]);
      const float e1 = __expf(s1[r]);
      lpart[r] += e0 + e1;
      const int row = r + hi * 8;
      Pw[row * PST + li]      = (__bf16)e0;
      Pw[row * PST + 16 + li] = (__bf16)e1;
    }

    // P B-fragment (K=m 0..31, N=n): 16 contiguous m per lane.
    const __bf16* pr = Pw + li * PST + hi * 16;
    v16bf pb = join16(load8bf(pr), load8bf(pr + 8));

    // O^T accumulation: D[M=d, N=n] += V(16d x 32m) * P^T(32m x 16n).
#pragma unroll
    for (int dt = 0; dt < 4; ++dt) {
      const __bf16* vl = &Vt[cur][(dt * 16 + li) * VST];
      v16bf va = join16(load8bf(vl + hi * 16), load8bf(vl + hi * 16 + 8));
      accd[dt] = wmma_bf16(va, pb, accd[dt]);
    }

    // Next tile's DMA must have landed before anyone reads it; also guards
    // the buffer we just consumed against the copy issued next iteration.
    wait_async0();
    __syncthreads();
  }

  // Row-sum reduction across each 16-lane half (rows 0..7 in lanes<16, 8..15 above).
#pragma unroll
  for (int r = 0; r < 8; ++r) {
    float v = lpart[r];
    v += __shfl_xor(v, 1);
    v += __shfl_xor(v, 2);
    v += __shfl_xor(v, 4);
    v += __shfl_xor(v, 8);
    lpart[r] = v;
  }
  if (lane == 0) {
#pragma unroll
    for (int r = 0; r < 8; ++r) ls[w][r] = lpart[r];
  }
  if (lane == 16) {
#pragma unroll
    for (int r = 0; r < 8; ++r) ls[w][8 + r] = lpart[r];
  }
  const float denom = ls[w][li];   // same-wave LDS ops are in order

  const float g  = gamma[h];
  const float sA = (g / (1.f + g)) / denom;  // scales softmax-normalized O
  const float sB = 1.f / (1.f + g);          // scales yp
  const int n = nbase + li;

#pragma unroll
  for (int dt = 0; dt < 4; ++dt) {
#pragma unroll
    for (int r = 0; r < 8; ++r) {
      const int d = dt * 16 + hi * 8 + r;
      const float ypv = ypws[((size_t)(b * 64) + d) * NTOK + n];
      out[((size_t)b * 256 + h * 64 + d) * NTOK + n] = accd[dt][r] * sA + ypv * sB;
    }
  }
}

extern "C" void kernel_launch(void* const* d_in, const int* in_sizes, int n_in,
                              void* d_out, int out_size, void* d_ws, size_t ws_size,
                              hipStream_t stream) {
  (void)in_sizes; (void)n_in; (void)out_size; (void)ws_size;
  const float* x     = (const float*)d_in[0];
  const float* y     = (const float*)d_in[1];
  const float* Wq    = (const float*)d_in[2];
  const float* bq    = (const float*)d_in[3];
  const float* Wk    = (const float*)d_in[4];
  const float* bk    = (const float*)d_in[5];
  const float* Wv    = (const float*)d_in[6];
  const float* bv    = (const float*)d_in[7];
  const float* Wp    = (const float*)d_in[8];
  const float* gamma = (const float*)d_in[9];

  char* ws = (char*)d_ws;
  __bf16* qT   = (__bf16*)(ws);                     // 4 MiB (b,h,n,d) bf16
  __bf16* kT   = (__bf16*)(ws + (4u << 20));        // 4 MiB (b,h,m,d) bf16
  __bf16* vws  = (__bf16*)(ws + (8u << 20));        // 4 MiB (b,h,d,m) bf16
  float*  ypws = (float*)(ws + (12u << 20));        // 2 MiB (b,d,n)  f32

  mha_proj_kernel<<<128, 128, 0, stream>>>(x, y, Wq, bq, Wk, bk, Wv, bv, Wp,
                                           qT, kT, vws, ypws);
  mha_attn_kernel<<<512, 128, 0, stream>>>(qT, kT, vws, ypws, gamma,
                                           (float*)d_out);
}
